// SampleAttentionLayer_37056977830547
// MI455X (gfx1250) — matrix operand order repaired and checked by reading, not compiled, not run on hardware
//
#include <hip/hip_runtime.h>
#include <stdint.h>

#define DM   512
#define QB   4096
#define QA   1024
#define BATCH 8
#define NKEY 4096

typedef __attribute__((ext_vector_type(16))) __bf16 v16bf;
typedef __attribute__((ext_vector_type(8)))  float  v8f;

union Frag { unsigned int u[8]; v16bf v; };

struct alignas(8) US4 { unsigned short x, y, z, w; };

__device__ __forceinline__ unsigned short f2bf(float f) {
  unsigned int u = __float_as_uint(f);
  u += 0x7FFFu + ((u >> 16) & 1u);          // round-to-nearest-even
  return (unsigned short)(u >> 16);
}
__device__ __forceinline__ float bf2f(unsigned short h) {
  return __uint_as_float(((unsigned int)h) << 16);
}

// Async global->LDS 16-byte copy (per lane). VDST = LDS byte address,
// 64-bit vaddr, saddr = off. Tracked by ASYNCcnt.
__device__ __forceinline__ void async_g2l_b128(unsigned int lds_byte_off,
                                               const void* gaddr) {
  asm volatile("global_load_async_to_lds_b128 %0, %1, off"
               :: "v"(lds_byte_off), "v"(gaddr) : "memory");
}
__device__ __forceinline__ void wait_async0() {
  asm volatile("s_wait_asynccnt 0x0" ::: "memory");
}

// ---------------------------------------------------------------------------
// Templated bf16 WMMA GEMM with double-buffered LDS staging.
//   C[m,n] = alpha * sum_k A[m,k] * B(k,n)  (+ bias)
//   TA/TB: float (converted to bf16 on commit) or unsigned short (bf16)
//   B_IS_KN:  true  -> B stored [K][N] row-major (transpose into LDS)
//             false -> B stored [N][K] row-major (W[n][k]; direct copy)
//   BIAS_MODE: 0 none, 1 bias[n], 2 bias[m]
// bf16 operands in direct-copy layout use GLOBAL_LOAD_ASYNC_TO_LDS_B128.
// Block: 256 threads (8 waves). Block tile 128x128, K-tile 64.
// Wave (wm 0..3, wn 0..1) owns 32x64 of C = 2x4 wmma accumulators.
// ---------------------------------------------------------------------------
template<typename TA, typename TB, bool B_IS_KN, bool OUT_BF16, int BIAS_MODE>
__global__ void __launch_bounds__(256)
gemm_bf16_wmma(const void* Ap, const void* Bp, const float* bias, void* Cp,
               int M, int N, int K,
               long long sA, long long sB, long long sC, float alpha)
{
  constexpr int BM = 128, BN = 128, BK = 64, LT = BK + 8;   // stride 144B: 16B-aligned rows, conflict-free frags
  constexpr bool ASY_A = (sizeof(TA) == 2);                 // A is always M-major
  constexpr bool ASY_B = (sizeof(TB) == 2) && !B_IS_KN;     // direct-copy B only
  constexpr bool ANY_ASYNC = ASY_A || ASY_B;

  __shared__ alignas(16) unsigned short As[2][BM * LT];
  __shared__ alignas(16) unsigned short Bs[2][BN * LT];     // Bt[n][k]

  const TA* A = (const TA*)Ap + (long long)blockIdx.z * sA;
  const TB* B = (const TB*)Bp + (long long)blockIdx.z * sB;

  const int tid  = threadIdx.x;
  const int lane = tid & 31;
  const int wave = tid >> 5;
  const int wm   = wave >> 1;          // 0..3
  const int wn   = wave & 1;           // 0..1
  const int bm0  = blockIdx.y * BM;
  const int bn0  = blockIdx.x * BN;
  const int l    = lane & 15;
  const int h    = lane >> 4;

  v8f acc[2][4] = {};

  // prefetch registers for the synchronous (conversion / transpose) paths
  float4 pfa[8];
  float4 pfb[8];
  US4    phb[8];

  // ---- A staging ----------------------------------------------------------
  auto issueA = [&](int buf, int k0) {
    if constexpr (ASY_A) {
      #pragma unroll
      for (int it = 0; it < (BM * BK) / (256 * 8); ++it) {        // 4 x b128
        int i = tid * 8 + it * 2048;
        int r = i >> 6, c = i & 63;
        unsigned lds = (unsigned)(size_t)&As[buf][r * LT + c];
        const void* g = (const unsigned short*)A +
                        (long long)(bm0 + r) * K + (k0 + c);
        async_g2l_b128(lds, g);
      }
    } else {
      #pragma unroll
      for (int it = 0; it < (BM * BK) / (256 * 4); ++it) {        // 8 x float4
        int i = tid * 4 + it * 1024;
        int r = i >> 6, c = i & 63;
        pfa[it] = *(const float4*)((const float*)A +
                                   (long long)(bm0 + r) * K + (k0 + c));
      }
    }
  };
  auto commitA = [&](int buf) {
    if constexpr (!ASY_A) {
      #pragma unroll
      for (int it = 0; it < (BM * BK) / (256 * 4); ++it) {
        int i = tid * 4 + it * 1024;
        int r = i >> 6, c = i & 63;
        US4 o = { f2bf(pfa[it].x), f2bf(pfa[it].y),
                  f2bf(pfa[it].z), f2bf(pfa[it].w) };
        *(US4*)&As[buf][r * LT + c] = o;
      }
    }
  };

  // ---- B staging ----------------------------------------------------------
  auto issueB = [&](int buf, int k0) {
    if constexpr (ASY_B) {
      #pragma unroll
      for (int it = 0; it < (BN * BK) / (256 * 8); ++it) {
        int i = tid * 8 + it * 2048;
        int r = i >> 6, c = i & 63;
        unsigned lds = (unsigned)(size_t)&Bs[buf][r * LT + c];
        const void* g = (const unsigned short*)B +
                        (long long)(bn0 + r) * K + (k0 + c);
        async_g2l_b128(lds, g);
      }
    } else if constexpr (!B_IS_KN) {                 // f32 W[n][k], direct
      #pragma unroll
      for (int it = 0; it < (BN * BK) / (256 * 4); ++it) {
        int i = tid * 4 + it * 1024;
        int r = i >> 6, c = i & 63;
        pfb[it] = *(const float4*)((const float*)B +
                                   (long long)(bn0 + r) * K + (k0 + c));
      }
    } else {                                         // B[k][n], transpose
      #pragma unroll
      for (int it = 0; it < (BK * BN) / (256 * 4); ++it) {
        int i = tid * 4 + it * 1024;
        int r = i >> 7, c = i & 127;                 // r = k, c = n
        const TB* src = B + (long long)(k0 + r) * N + (bn0 + c);
        if constexpr (sizeof(TB) == 4) pfb[it] = *(const float4*)src;
        else                           phb[it] = *(const US4*)src;
      }
    }
  };
  auto commitB = [&](int buf) {
    if constexpr (ASY_B) {
      // nothing: async writes LDS directly
    } else if constexpr (!B_IS_KN) {
      #pragma unroll
      for (int it = 0; it < (BN * BK) / (256 * 4); ++it) {
        int i = tid * 4 + it * 1024;
        int r = i >> 6, c = i & 63;
        US4 o = { f2bf(pfb[it].x), f2bf(pfb[it].y),
                  f2bf(pfb[it].z), f2bf(pfb[it].w) };
        *(US4*)&Bs[buf][r * LT + c] = o;
      }
    } else {
      #pragma unroll
      for (int it = 0; it < (BK * BN) / (256 * 4); ++it) {
        int i = tid * 4 + it * 1024;
        int r = i >> 7, c = i & 127;
        if constexpr (sizeof(TB) == 4) {
          Bs[buf][(c + 0) * LT + r] = f2bf(pfb[it].x);
          Bs[buf][(c + 1) * LT + r] = f2bf(pfb[it].y);
          Bs[buf][(c + 2) * LT + r] = f2bf(pfb[it].z);
          Bs[buf][(c + 3) * LT + r] = f2bf(pfb[it].w);
        } else {
          Bs[buf][(c + 0) * LT + r] = phb[it].x;
          Bs[buf][(c + 1) * LT + r] = phb[it].y;
          Bs[buf][(c + 2) * LT + r] = phb[it].z;
          Bs[buf][(c + 3) * LT + r] = phb[it].w;
        }
      }
    }
  };

  // ---- compute: two wmma k-steps over a 64-wide tile ----------------------
  auto compute = [&](int cur) {
    #pragma unroll
    for (int ks = 0; ks < 2; ++ks) {
      const int kb = ks * 32;
      Frag a[2], b[4];
      // A fragment (16x32): lane l = row, k = (j<4 ? 2j : 16+2(j-4)) + 8h
      #pragma unroll
      for (int mt = 0; mt < 2; ++mt) {
        const unsigned short* base = &As[cur][(wm * 32 + mt * 16 + l) * LT + kb];
        #pragma unroll
        for (int j = 0; j < 8; ++j) {
          int k = ((j < 4) ? (2 * j) : (16 + 2 * (j - 4))) + 8 * h;
          a[mt].u[j] = *(const unsigned int*)(base + k);
        }
      }
      // B fragment (32x16): lane l = column n, k = (j<4 ? 2j : 16+2(j-4)) + 8h (from Bt[n][k])
      #pragma unroll
      for (int nt = 0; nt < 4; ++nt) {
        const unsigned short* base = &Bs[cur][(wn * 64 + nt * 16 + l) * LT + kb];
        #pragma unroll
        for (int j = 0; j < 8; ++j) {
          int k = ((j < 4) ? (2 * j) : (16 + 2 * (j - 4))) + 8 * h;
          b[nt].u[j] = *(const unsigned int*)(base + k);
        }
      }
      #pragma unroll
      for (int mt = 0; mt < 2; ++mt)
        #pragma unroll
        for (int nt = 0; nt < 4; ++nt)
          acc[mt][nt] = __builtin_amdgcn_wmma_f32_16x16x32_bf16(
              false, a[mt].v, false, b[nt].v, (short)0, acc[mt][nt],
              false, false);
    }
  };

  // ---- double-buffered pipeline -------------------------------------------
  const int nkt = K / BK;
  issueA(0, 0); issueB(0, 0);
  commitA(0);   commitB(0);
  if constexpr (ANY_ASYNC) wait_async0();
  __syncthreads();

  for (int t = 0; t < nkt; ++t) {
    const int cur = t & 1, nxt = cur ^ 1;
    const bool more = (t + 1 < nkt);
    if (more) { issueA(nxt, (t + 1) * BK); issueB(nxt, (t + 1) * BK); }
    compute(cur);
    if (more) {
      commitA(nxt); commitB(nxt);
      if constexpr (ANY_ASYNC) wait_async0();
      __syncthreads();
    }
  }

  // ---- epilogue: C layout => VGPR v: row = v + 8h, col = lane&15 ----------
  #pragma unroll
  for (int mt = 0; mt < 2; ++mt) {
    #pragma unroll
    for (int nt = 0; nt < 4; ++nt) {
      const int gr0 = bm0 + wm * 32 + mt * 16 + 8 * h;
      const int gc  = bn0 + wn * 64 + nt * 16 + l;
      float bn_ = 0.0f;
      if constexpr (BIAS_MODE == 1) bn_ = bias[gc];
      #pragma unroll
      for (int v = 0; v < 8; ++v) {
        const int gr = gr0 + v;
        float val = acc[mt][nt][v] * alpha;
        if constexpr (BIAS_MODE == 1) val += bn_;
        if constexpr (BIAS_MODE == 2) val += bias[gr];
        if constexpr (OUT_BF16) {
          unsigned short* C = (unsigned short*)Cp + (long long)blockIdx.z * sC;
          C[(long long)gr * N + gc] = f2bf(val);
        } else {
          float* C = (float*)Cp + (long long)blockIdx.z * sC;
          C[(long long)gr * N + gc] = val;
        }
      }
    }
  }
}

// ---------------------------------------------------------------------------
// In-place masked softmax over rows of S (bf16), row length = NKEY.
// One block (256 threads) per (b, q) row.
// ---------------------------------------------------------------------------
__global__ void __launch_bounds__(256)
softmax_mask_kernel(unsigned short* S, const unsigned char* mask)
{
  __shared__ float redmax[8];
  __shared__ float redsum[8];
  const long long row = blockIdx.x;                  // b*QA + q
  unsigned short*      srow = S    + row * (long long)NKEY;
  const unsigned char* mrow = mask + row * (long long)NKEY;
  const int tid = threadIdx.x;

  float v[NKEY / 256];
  float vmax = -__builtin_inff();
  #pragma unroll
  for (int i = 0; i < NKEY / 256; ++i) {
    int idx = tid + i * 256;
    float x = mrow[idx] ? -__builtin_inff() : bf2f(srow[idx]);
    v[i] = x;
    vmax = fmaxf(vmax, x);
  }
  #pragma unroll
  for (int o = 16; o > 0; o >>= 1) vmax = fmaxf(vmax, __shfl_xor(vmax, o, 32));
  if ((tid & 31) == 0) redmax[tid >> 5] = vmax;
  __syncthreads();
  float m = redmax[0];
  #pragma unroll
  for (int w = 1; w < 8; ++w) m = fmaxf(m, redmax[w]);

  float sum = 0.0f;
  #pragma unroll
  for (int i = 0; i < NKEY / 256; ++i) {
    float e = __expf(v[i] - m);
    v[i] = e;
    sum += e;
  }
  #pragma unroll
  for (int o = 16; o > 0; o >>= 1) sum += __shfl_xor(sum, o, 32);
  if ((tid & 31) == 0) redsum[tid >> 5] = sum;
  __syncthreads();
  float tot = 0.0f;
  #pragma unroll
  for (int w = 0; w < 8; ++w) tot += redsum[w];
  const float inv = 1.0f / tot;

  #pragma unroll
  for (int i = 0; i < NKEY / 256; ++i)
    srow[tid + i * 256] = f2bf(v[i] * inv);
}

// ---------------------------------------------------------------------------
extern "C" void kernel_launch(void* const* d_in, const int* in_sizes, int n_in,
                              void* d_out, int out_size, void* d_ws, size_t ws_size,
                              hipStream_t stream)
{
  (void)in_sizes; (void)n_in; (void)out_size; (void)ws_size;
  const float* queries = (const float*)d_in[0];   // [8,4096,512]
  const float* keys    = (const float*)d_in[1];   // [8,4096,512]
  const float* values  = (const float*)d_in[2];   // [8,4096,512]
  const unsigned char* mask = (const unsigned char*)d_in[3]; // [8,1024,4096] bool
  const float* Wq = (const float*)d_in[4];
  const float* bq = (const float*)d_in[5];
  const float* Wk = (const float*)d_in[6];
  const float* bk = (const float*)d_in[7];
  const float* Wv = (const float*)d_in[8];
  const float* bv = (const float*)d_in[9];
  const float* Wo = (const float*)d_in[10];
  const float* bo = (const float*)d_in[11];
  const float* Wd = (const float*)d_in[12];       // [1024,4096]
  const float* bd = (const float*)d_in[13];       // [1024]

  // workspace layout (bf16 intermediates, ~144 MB total)
  unsigned short* Xq = (unsigned short*)d_ws;                 // [8,1024,512]
  unsigned short* Qb = Xq + (size_t)BATCH * QA * DM;          // [8,1024,512]
  unsigned short* Kb = Qb + (size_t)BATCH * QA * DM;          // [8,4096,512]
  unsigned short* Vb = Kb + (size_t)BATCH * NKEY * DM;        // [8,4096,512]
  unsigned short* Sb = Vb + (size_t)BATCH * NKEY * DM;        // [8,1024,4096]
  unsigned short* Ob = Xq;  // Xq dead after Q projection -> reuse for O

  const dim3 blk(256);
  const float rs = 0.04419417382415922f;   // 1/sqrt(512)

  // 1) Downsample: Xq[b,qa,d] = sum_qb Wd[qa,qb]*queries[b,qb,d] + bd[qa]
  gemm_bf16_wmma<float, float, true, true, 2>
      <<<dim3(DM / 128, QA / 128, BATCH), blk, 0, stream>>>(
      Wd, queries, bd, Xq, QA, DM, QB,
      0LL, (long long)QB * DM, (long long)QA * DM, 1.0f);

  // 2) Q = Xq @ Wq^T + bq      (A bf16 async, B = W NK layout, bias on N)
  gemm_bf16_wmma<unsigned short, float, false, true, 1>
      <<<dim3(DM / 128, QA / 128, BATCH), blk, 0, stream>>>(
      Xq, Wq, bq, Qb, QA, DM, DM,
      (long long)QA * DM, 0LL, (long long)QA * DM, 1.0f);

  // 3) K = keys @ Wk^T + bk
  gemm_bf16_wmma<float, float, false, true, 1>
      <<<dim3(DM / 128, NKEY / 128, BATCH), blk, 0, stream>>>(
      keys, Wk, bk, Kb, NKEY, DM, DM,
      (long long)NKEY * DM, 0LL, (long long)NKEY * DM, 1.0f);

  // 4) V = values @ Wv^T + bv
  gemm_bf16_wmma<float, float, false, true, 1>
      <<<dim3(DM / 128, NKEY / 128, BATCH), blk, 0, stream>>>(
      values, Wv, bv, Vb, NKEY, DM, DM,
      (long long)NKEY * DM, 0LL, (long long)NKEY * DM, 1.0f);

  // 5) S = (Q @ K^T) / sqrt(512)   (A and B bf16 -> both async staged)
  gemm_bf16_wmma<unsigned short, unsigned short, false, true, 0>
      <<<dim3(NKEY / 128, QA / 128, BATCH), blk, 0, stream>>>(
      Qb, Kb, nullptr, Sb, QA, NKEY, DM,
      (long long)QA * DM, (long long)NKEY * DM, (long long)QA * NKEY, rs);

  // 6) masked softmax (in place, S -> P)
  softmax_mask_kernel<<<dim3(BATCH * QA), blk, 0, stream>>>(Sb, mask);

  // 7) O = P @ V               (A = P bf16 async; B = V in KN layout)
  gemm_bf16_wmma<unsigned short, unsigned short, true, true, 0>
      <<<dim3(DM / 128, QA / 128, BATCH), blk, 0, stream>>>(
      Sb, Vb, nullptr, Ob, QA, DM, NKEY,
      (long long)QA * NKEY, (long long)NKEY * DM, (long long)QA * DM, 1.0f);

  // 8) out = O @ Wo^T + bo     (f32 output to d_out)
  gemm_bf16_wmma<unsigned short, float, false, false, 1>
      <<<dim3(DM / 128, QA / 128, BATCH), blk, 0, stream>>>(
      Ob, Wo, bo, d_out, QA, DM, DM,
      (long long)QA * DM, 0LL, (long long)QA * DM, 1.0f);
}
